// co_non_local_block_91139206021612
// MI455X (gfx1250) — matrix operand order repaired and checked by reading, not compiled, not run on hardware
//
#include <hip/hip_runtime.h>

typedef __attribute__((ext_vector_type(16))) _Float16 v16h;
typedef __attribute__((ext_vector_type(8)))  float    v8f;

#define HW 4096
#define WD 64
#define CH 64
#define RR 32

union Frag { v16h h; float4 f[2]; };

// ---------------- projection: theta (hw,32 f16), phiT (hw,32 f16), g (32,hw f16) ----------------
__global__ void proj_kernel(const float* __restrict__ left, const float* __restrict__ right,
                            const float* __restrict__ pre_l, const float* __restrict__ pre_r,
                            const float* __restrict__ query_l, const float* __restrict__ key_l,
                            const float* __restrict__ query_r, const float* __restrict__ key_r,
                            const float* __restrict__ w_theta, const float* __restrict__ b_theta,
                            const float* __restrict__ w_phi,   const float* __restrict__ b_phi,
                            const float* __restrict__ w_g,     const float* __restrict__ b_g,
                            _Float16* __restrict__ thetaH, _Float16* __restrict__ phiH,
                            _Float16* __restrict__ gH) {
  int idx = blockIdx.x * blockDim.x + threadIdx.x;   // 4 * 4096 * 32 threads exactly
  int r  = idx & 31;
  int q  = (idx >> 5) & (HW - 1);
  int sb = idx >> 17;                                 // 0..3  (side*2 + batch)
  int s = sb >> 1, b = sb & 1;
  const float* xq   = s ? right   : left;
  const float* xk   = s ? left    : right;
  const float* pre  = s ? pre_r   : pre_l;
  const float* qadd = s ? query_r : query_l;
  const float* kadd = s ? key_l   : key_r;

  float at = b_theta[r], ap = b_phi[r], ag = b_g[r];
  for (int c = 0; c < CH; ++c) {
    float xv = xq[(b * CH + c) * HW + q];
    float kv = xk[(b * CH + c) * HW + q];
    at += w_theta[r * (CH + 1) + c] * xv;
    ap += w_phi[r * CH + c] * kv;
    ag += w_g[r * CH + c] * kv;
  }
  at += w_theta[r * (CH + 1) + CH] * (pre[b * HW + q] * (1.0f / WD));  // pre / w channel
  at += qadd[(b * RR + r) * HW + q];
  ap += kadd[(b * RR + r) * HW + q];

  size_t base = (size_t)sb * HW * RR;
  thetaH[base + (size_t)q * RR + r] = (_Float16)at;   // (q, r) row-major
  phiH  [base + (size_t)q * RR + r] = (_Float16)ap;   // phi^T: (k, r) row-major
  gH    [base + (size_t)r * HW + q] = (_Float16)ag;   // (r, k) row-major
}

// ---------------- fused attention: S = theta*phi, exp, P*g^T, exp-col, all via WMMA ----------------
__global__ void __launch_bounds__(256, 1)
attn_kernel(const _Float16* __restrict__ thetaH, const _Float16* __restrict__ phiH,
            const _Float16* __restrict__ gH, float* __restrict__ aout,
            float* __restrict__ dout) {
  __shared__ __align__(16) _Float16 lds[8][16 * 32];   // per-wave P staging tile

  int lane = threadIdx.x & 31;
  int wv   = threadIdx.x >> 5;
  int gw   = blockIdx.x * 8 + wv;     // 0..1023 wave tiles
  int sb   = gw >> 8;                 // 0..3
  int qt   = gw & 255;
  int q0   = qt * 16;
  int s = sb >> 1, b = sb & 1;

  const _Float16* th = thetaH + (size_t)sb * HW * RR;
  const _Float16* ph = phiH   + (size_t)sb * HW * RR;
  const _Float16* gm = gH     + (size_t)sb * HW * RR;

  int ln = lane & 15;           // M for A, N for B/C
  int kb = (lane >> 4) * 8;     // unused; A fragments also load at nb (contiguous 16)
  int nb = (lane >> 4) * 16;    // B-fragment K base (contiguous 16)
  int rowh = (lane >> 4) * 8;   // C-fragment row base

  Frag aT;
  aT.f[0] = *reinterpret_cast<const float4*>(th + (size_t)(q0 + ln) * RR + nb);
  aT.f[1] = *reinterpret_cast<const float4*>(th + (size_t)(q0 + ln) * RR + nb + 8);

  v8f o0 = {}, o1 = {};
  float dsum[8], ex[8];
#pragma unroll
  for (int v = 0; v < 8; ++v) { dsum[v] = 0.f; ex[v] = 0.f; }
  _Float16* pl = &lds[wv][0];

  for (int k0 = 0; k0 < HW; k0 += 32) {
    Frag b0, b1;
    b0.f[0] = *reinterpret_cast<const float4*>(ph + (size_t)(k0 + ln) * RR + nb);
    b0.f[1] = *reinterpret_cast<const float4*>(ph + (size_t)(k0 + ln) * RR + nb + 8);
    b1.f[0] = *reinterpret_cast<const float4*>(ph + (size_t)(k0 + 16 + ln) * RR + nb);
    b1.f[1] = *reinterpret_cast<const float4*>(ph + (size_t)(k0 + 16 + ln) * RR + nb + 8);

    v8f s0 = {}, s1 = {};
    s0 = __builtin_amdgcn_wmma_f32_16x16x32_f16(false, aT.h, false, b0.h, (short)0, s0, false, false);
    s1 = __builtin_amdgcn_wmma_f32_16x16x32_f16(false, aT.h, false, b1.h, (short)0, s1, false, false);

    float c0 = (float)((k0 + ln) & (WD - 1));        // col % W for keys k0+ln
    float c1 = (float)((k0 + 16 + ln) & (WD - 1));
#pragma unroll
    for (int v = 0; v < 8; ++v) {
      float p0 = __expf(s0[v]);
      float p1 = __expf(s1[v]);
      dsum[v] += p0 + p1;
      ex[v]   += p0 * c0 + p1 * c1;
      int row = v + rowh;
      pl[row * 32 + ln]      = (_Float16)p0;
      pl[row * 32 + 16 + ln] = (_Float16)p1;
    }
    asm volatile("s_wait_dscnt 0" ::: "memory");     // in-wave LDS store->load hazard

    Frag aP, g0, g1;
    aP.f[0] = *reinterpret_cast<const float4*>(pl + ln * 32 + nb);
    aP.f[1] = *reinterpret_cast<const float4*>(pl + ln * 32 + nb + 8);
    g0.f[0] = *reinterpret_cast<const float4*>(gm + (size_t)ln * HW + k0 + nb);
    g0.f[1] = *reinterpret_cast<const float4*>(gm + (size_t)ln * HW + k0 + nb + 8);
    g1.f[0] = *reinterpret_cast<const float4*>(gm + (size_t)(16 + ln) * HW + k0 + nb);
    g1.f[1] = *reinterpret_cast<const float4*>(gm + (size_t)(16 + ln) * HW + k0 + nb + 8);

    o0 = __builtin_amdgcn_wmma_f32_16x16x32_f16(false, aP.h, false, g0.h, (short)0, o0, false, false);
    o1 = __builtin_amdgcn_wmma_f32_16x16x32_f16(false, aP.h, false, g1.h, (short)0, o1, false, false);
    asm volatile("s_wait_dscnt 0" ::: "memory");     // keep next-iter stores behind these loads
  }

  // reduce Sum(exp) and Sum(exp*col) across the 16 lanes sharing each row group
#pragma unroll
  for (int m = 1; m < 16; m <<= 1) {
#pragma unroll
    for (int v = 0; v < 8; ++v) {
      dsum[v] += __shfl_xor(dsum[v], m, 32);
      ex[v]   += __shfl_xor(ex[v],   m, 32);
    }
  }

  float* ao = aout + (size_t)sb * RR * HW;           // a_out: (r, q) row-major per (s,b)
#pragma unroll
  for (int v = 0; v < 8; ++v) {
    int q = q0 + v + rowh;
    float inv = 1.0f / dsum[v];
    ao[(size_t)ln * HW + q]        = o0[v] * inv;    // r = ln
    ao[(size_t)(16 + ln) * HW + q] = o1[v] * inv;    // r = 16 + ln
  }
  if (ln == 0) {                                     // disparity index output
    float* idxo = dout + 1048576 + s * 8192 + b * HW;
#pragma unroll
    for (int v = 0; v < 8; ++v) {
      int q = q0 + v + rowh;
      idxo[q] = (float)(q & (WD - 1)) - ex[v] / dsum[v];
    }
  }
}

// ---------------- epilogue 1: y = conv1x1(a, w_up)+b_up ; per-channel batch stats ----------------
__global__ void epi1_kernel(const float* __restrict__ aout, const float* __restrict__ w_up,
                            const float* __restrict__ b_up, float* __restrict__ ybuf,
                            float* __restrict__ stats) {
  __shared__ float rs[256], rq[256];
  int s = blockIdx.x >> 6;
  int o = blockIdx.x & 63;
  int tid = threadIdx.x;
  float wl[RR];
#pragma unroll
  for (int r = 0; r < RR; ++r) wl[r] = w_up[o * RR + r];
  float bu = b_up[o];
  float sum = 0.f, sq = 0.f;
  for (int i = tid; i < 2 * HW; i += 256) {
    int b = i >> 12;
    int q = i & (HW - 1);
    const float* a = aout + (size_t)(s * 2 + b) * RR * HW;
    float y = bu;
#pragma unroll
    for (int r = 0; r < RR; ++r) y += wl[r] * a[(size_t)r * HW + q];
    ybuf[((size_t)(s * 2 + b) * CH + o) * HW + q] = y;
    sum += y; sq += y * y;
  }
  rs[tid] = sum; rq[tid] = sq; __syncthreads();
  for (int st = 128; st > 0; st >>= 1) {
    if (tid < st) { rs[tid] += rs[tid + st]; rq[tid] += rq[tid + st]; }
    __syncthreads();
  }
  if (tid == 0) {
    float mean = rs[0] / (float)(2 * HW);
    float var  = rq[0] / (float)(2 * HW) - mean * mean;   // biased var
    stats[(s * CH + o) * 2]     = mean;
    stats[(s * CH + o) * 2 + 1] = rsqrtf(var + 1e-5f);
  }
}

// ---------------- epilogue 2: out = x + BN(y) ----------------
__global__ void epi2_kernel(const float* __restrict__ left, const float* __restrict__ right,
                            const float* __restrict__ ybuf, const float* __restrict__ stats,
                            const float* __restrict__ gamma, const float* __restrict__ beta,
                            float* __restrict__ dout) {
  int idx = blockIdx.x * blockDim.x + threadIdx.x;   // 1<<20 threads exactly
  int s   = idx >> 19;
  int rem = idx & ((1 << 19) - 1);                   // b*262144 + o*4096 + q
  int o   = (rem >> 12) & 63;
  const float* x = s ? right : left;
  float y    = ybuf[(size_t)s * (1 << 19) + rem];
  float mean = stats[(s * CH + o) * 2];
  float inv  = stats[(s * CH + o) * 2 + 1];
  dout[(size_t)s * (1 << 19) + rem] = x[rem] + (y - mean) * inv * gamma[o] + beta[o];
}

extern "C" void kernel_launch(void* const* d_in, const int* in_sizes, int n_in,
                              void* d_out, int out_size, void* d_ws, size_t ws_size,
                              hipStream_t stream) {
  const float* left    = (const float*)d_in[0];
  const float* right   = (const float*)d_in[1];
  const float* pre_l   = (const float*)d_in[2];
  const float* pre_r   = (const float*)d_in[3];
  const float* query_l = (const float*)d_in[4];
  const float* key_l   = (const float*)d_in[5];
  const float* query_r = (const float*)d_in[6];
  const float* key_r   = (const float*)d_in[7];
  const float* w_theta = (const float*)d_in[8];
  const float* b_theta = (const float*)d_in[9];
  const float* w_phi   = (const float*)d_in[10];
  const float* b_phi   = (const float*)d_in[11];
  const float* w_g     = (const float*)d_in[12];
  const float* b_g     = (const float*)d_in[13];
  const float* w_up    = (const float*)d_in[14];
  const float* b_up    = (const float*)d_in[15];
  const float* gamma   = (const float*)d_in[16];
  const float* beta    = (const float*)d_in[17];

  char* ws = (char*)d_ws;
  _Float16* thetaH = (_Float16*)(ws + 0);                 // 1 MB
  _Float16* phiH   = (_Float16*)(ws + (1u << 20));        // 1 MB
  _Float16* gH     = (_Float16*)(ws + (2u << 20));        // 1 MB
  float*    aout   = (float*)(ws + (3u << 20));           // 2 MB
  float*    ybuf   = (float*)(ws + (5u << 20));           // 4 MB
  float*    stats  = (float*)(ws + (9u << 20));           // 1 KB

  float* out = (float*)d_out;

  proj_kernel<<<2048, 256, 0, stream>>>(left, right, pre_l, pre_r,
                                        query_l, key_l, query_r, key_r,
                                        w_theta, b_theta, w_phi, b_phi, w_g, b_g,
                                        thetaH, phiH, gH);
  attn_kernel<<<128, 256, 0, stream>>>(thetaH, phiH, gH, aout, out);
  epi1_kernel<<<128, 256, 0, stream>>>(aout, w_up, b_up, ybuf, stats);
  epi2_kernel<<<4096, 256, 0, stream>>>(left, right, ybuf, stats, gamma, beta, out);
}
